// DecoderCellContent_76605036691810
// MI455X (gfx1250) — compile-verified, matrix operand order checked by reading, not run on hardware
//
#include <hip/hip_runtime.h>
#include <math.h>

// ---- dims ----
#define Bdim 64
#define Tdim 256
#define Ndim 256
#define Edim 512
#define Sdim 256
#define Hdim 512
#define Adim 256
#define EMBdim 256
#define Vdim 2048
#define INdim (Edim + EMBdim)   // 768

typedef __attribute__((ext_vector_type(16))) __bf16 v16bf;
typedef __attribute__((ext_vector_type(8)))  float  v8f;

// ---------------------------------------------------------------------------
// WMMA operand gathers per CDNA5 ISA 7.12.2 (16-bit A 16x32 layout; B mirrors
// with lane<->N; C/D: m = vgpr + 8*half, n = lane&15).
// Per-lane accesses are two contiguous 16B runs -> compiler fuses to b128.
// ---------------------------------------------------------------------------
__device__ __forceinline__ int wmma_kmap(int v, int half) {
  return (v < 4) ? (half * 8 + v * 2) : (16 + half * 8 + (v - 4) * 2);
}

__device__ __forceinline__ v16bf load_a_tile(const __bf16* a, int lda, int m0, int k0, int lane) {
  const int m = m0 + (lane & 15);
  const int half = lane >> 4;
  v16bf r;
#pragma unroll
  for (int v = 0; v < 8; ++v) {
    const int k = k0 + wmma_kmap(v, half);
    r[2 * v]     = a[(size_t)m * lda + k];
    r[2 * v + 1] = a[(size_t)m * lda + k + 1];
  }
  return r;
}

// NK == true : B stored [N, K] row-major (torch W[out,in], computing X @ W^T)
// NK == false: B stored [K, N] row-major (computing X @ W)
template <bool NK>
__device__ __forceinline__ v16bf load_b_tile(const __bf16* b, int ldb, int k0, int n0, int lane) {
  const int n = n0 + (lane & 15);
  const int half = lane >> 4;
  v16bf r;
#pragma unroll
  for (int v = 0; v < 8; ++v) {
    const int k = k0 + wmma_kmap(v, half);
    if (NK) {
      r[2 * v]     = b[(size_t)n * ldb + k];
      r[2 * v + 1] = b[(size_t)n * ldb + k + 1];
    } else {
      r[2 * v]     = b[(size_t)k * ldb + n];
      r[2 * v + 1] = b[(size_t)(k + 1) * ldb + n];
    }
  }
  return r;
}

// Generic bf16 GEMM: C[M,N] = A[M,K] (row-major bf16) * B + bias (f32 acc).
// One wave owns a 16x64 output panel: A tile loaded once per k-step, reused
// across 4 B tiles -> 4 independent v_wmma_f32_16x16x32_bf16 back to back.
template <bool NK>
__global__ void gemm_bf16_kernel(const __bf16* __restrict__ A, const __bf16* __restrict__ B,
                                 const float* __restrict__ bias, float* __restrict__ C,
                                 int M, int N, int K, int ldb) {
  const int wave = (blockIdx.x * blockDim.x + threadIdx.x) >> 5;
  const int lane = threadIdx.x & 31;
  const int ntn = N >> 6;                   // 64-wide N panels
  const int tiles = (M >> 4) * ntn;
  if (wave >= tiles) return;                // wave-uniform: EXEC all-ones for WMMA
  const int mt = wave / ntn;
  const int nb = (wave - mt * ntn) << 6;    // panel start column

  v8f acc0 = {}, acc1 = {}, acc2 = {}, acc3 = {};
  for (int k0 = 0; k0 < K; k0 += 32) {
    // prefetch next k-slice of the L2-resident operands toward the WGP
    if (k0 + 32 < K) {
      __builtin_prefetch(&A[(size_t)(mt * 16 + (lane & 15)) * K + k0 + 32], 0, 0);
      if (NK)
        __builtin_prefetch(&B[(size_t)(nb + (lane & 15)) * ldb + k0 + 32], 0, 0);
      else
        __builtin_prefetch(&B[(size_t)(k0 + 32 + (lane & 15)) * ldb + nb], 0, 0);
    }
    v16bf a  = load_a_tile(A, K, mt * 16, k0, lane);
    v16bf b0 = load_b_tile<NK>(B, ldb, k0, nb +  0, lane);
    v16bf b1 = load_b_tile<NK>(B, ldb, k0, nb + 16, lane);
    v16bf b2 = load_b_tile<NK>(B, ldb, k0, nb + 32, lane);
    v16bf b3 = load_b_tile<NK>(B, ldb, k0, nb + 48, lane);
    acc0 = __builtin_amdgcn_wmma_f32_16x16x32_bf16(false, a, false, b0, (short)0, acc0, false, false);
    acc1 = __builtin_amdgcn_wmma_f32_16x16x32_bf16(false, a, false, b1, (short)0, acc1, false, false);
    acc2 = __builtin_amdgcn_wmma_f32_16x16x32_bf16(false, a, false, b2, (short)0, acc2, false, false);
    acc3 = __builtin_amdgcn_wmma_f32_16x16x32_bf16(false, a, false, b3, (short)0, acc3, false, false);
  }

  const int nl = lane & 15;
  const int half = lane >> 4;
  v8f accs[4] = {acc0, acc1, acc2, acc3};
#pragma unroll
  for (int j = 0; j < 4; ++j) {
    const int n = nb + j * 16 + nl;
    const float bv = bias ? bias[n] : 0.0f;
#pragma unroll
    for (int v = 0; v < 8; ++v) {
      const int m = mt * 16 + v + 8 * half;
      C[(size_t)m * N + n] = accs[j][v] + bv;
    }
  }
}

// ---------------------------------------------------------------------------
// Elementwise / reduction kernels
// ---------------------------------------------------------------------------
__global__ void f32_to_bf16_kernel(const float* __restrict__ src, __bf16* __restrict__ dst, int n) {
  int i = blockIdx.x * blockDim.x + threadIdx.x;
  if (i < n) dst[i] = (__bf16)src[i];
}

__global__ void init_state_kernel(float* __restrict__ h, __bf16* __restrict__ h_bf,
                                  float* __restrict__ loss_acc) {
  int i = blockIdx.x * blockDim.x + threadIdx.x;
  if (i < Bdim * Hdim) { h[i] = 0.0f; h_bf[i] = (__bf16)0.0f; }
  if (i == 0) *loss_acc = 0.0f;
}

// scores = tanh(encp[b,n,:] + sproj[b,:] + cproj[b,:]) . v_att + b_att ; softmax over n;
// context[b,:] = alpha . enc[b,:,:]; write context (bf16) into x[:, 0:512].
__global__ void att_softmax_ctx_kernel(const float* __restrict__ encp,
                                       const float* __restrict__ sproj,
                                       const float* __restrict__ cproj,
                                       const float* __restrict__ v_att,
                                       const float* __restrict__ b_att,
                                       const float* __restrict__ enc,
                                       __bf16* __restrict__ x_bf) {
  const int b = blockIdx.x;
  const int tid = threadIdx.x;            // 256 threads; tid == pixel n
  __shared__ float svc[Adim];
  __shared__ float vat[Adim];
  __shared__ float red[Ndim];
  __shared__ float alpha[Ndim];

  svc[tid] = sproj[b * Adim + tid] + cproj[b * Adim + tid];
  vat[tid] = v_att[tid];
  __syncthreads();

  const float* ep = encp + ((size_t)b * Ndim + tid) * Adim;
  float s = b_att[0];
#pragma unroll 4
  for (int a = 0; a < Adim; ++a) s += tanhf(ep[a] + svc[a]) * vat[a];

  red[tid] = s; __syncthreads();
  for (int off = 128; off > 0; off >>= 1) {
    if (tid < off) red[tid] = fmaxf(red[tid], red[tid + off]);
    __syncthreads();
  }
  const float mx = red[0]; __syncthreads();
  const float ex = expf(s - mx);
  red[tid] = ex; __syncthreads();
  for (int off = 128; off > 0; off >>= 1) {
    if (tid < off) red[tid] += red[tid + off];
    __syncthreads();
  }
  alpha[tid] = ex / red[0];
  __syncthreads();

  const float* eb = enc + (size_t)b * Ndim * Edim;
  for (int e = tid; e < Edim; e += 256) {
    float c = 0.0f;
#pragma unroll 4
    for (int n = 0; n < Ndim; ++n) c += alpha[n] * eb[(size_t)n * Edim + e];
    x_bf[b * INdim + e] = (__bf16)c;
  }
}

// x[:, 512:768] = embedding[token] (bf16); token = t==0 ? START : target[b, t-1]
__global__ void embed_kernel(const float* __restrict__ emb, const int* __restrict__ target,
                             int t, __bf16* __restrict__ x_bf) {
  int idx = blockIdx.x * blockDim.x + threadIdx.x;   // 64*256
  int b = idx >> 8, j = idx & 255;
  int tok = (t == 0) ? 0 : target[b * Tdim + (t - 1)];
  x_bf[b * INdim + Edim + j] = (__bf16)emb[(size_t)tok * EMBdim + j];
}

// GRU gates + state update (in-place h), also refresh bf16 copy of h.
__global__ void gru_elem_kernel(const float* __restrict__ gi, const float* __restrict__ gh,
                                float* __restrict__ h, __bf16* __restrict__ h_bf) {
  int idx = blockIdx.x * blockDim.x + threadIdx.x;   // 64*512
  int b = idx >> 9, j = idx & 511;
  const float* gib = gi + (size_t)b * 3 * Hdim;
  const float* ghb = gh + (size_t)b * 3 * Hdim;
  float r = 1.0f / (1.0f + expf(-(gib[j] + ghb[j])));
  float z = 1.0f / (1.0f + expf(-(gib[Hdim + j] + ghb[Hdim + j])));
  float n = tanhf(gib[2 * Hdim + j] + r * ghb[2 * Hdim + j]);
  float hv = h[idx];
  float hn = (1.0f - z) * n + z * hv;
  h[idx] = hn;
  h_bf[idx] = (__bf16)hn;
}

// cross-entropy over logits row (already written to d_out); accumulate -logp(tgt)
__global__ void loss_kernel(const float* __restrict__ logits_t, const int* __restrict__ target,
                            int t, float* __restrict__ loss_acc) {
  const int b = blockIdx.x;
  const int tid = threadIdx.x;             // 256
  const float* lg = logits_t + (size_t)b * Vdim;
  __shared__ float red[256];
  float m = -3.402823466e+38f;
  for (int v = tid; v < Vdim; v += 256) m = fmaxf(m, lg[v]);
  red[tid] = m; __syncthreads();
  for (int off = 128; off > 0; off >>= 1) {
    if (tid < off) red[tid] = fmaxf(red[tid], red[tid + off]);
    __syncthreads();
  }
  m = red[0]; __syncthreads();
  float s = 0.0f;
  for (int v = tid; v < Vdim; v += 256) s += expf(lg[v] - m);
  red[tid] = s; __syncthreads();
  for (int off = 128; off > 0; off >>= 1) {
    if (tid < off) red[tid] += red[tid + off];
    __syncthreads();
  }
  if (tid == 0) {
    int tgt = target[b * Tdim + t];
    float logp = lg[tgt] - m - logf(red[0]);
    atomicAdd(loss_acc, -logp);
  }
}

__global__ void final_loss_kernel(const float* __restrict__ loss_acc, float* __restrict__ out) {
  // reference: loss_t = mean over B; loss = sum_t / T / B  ->  sum/(T*B*B)
  out[0] = loss_acc[0] / ((float)Tdim * (float)Bdim * (float)Bdim);
}

// ---------------------------------------------------------------------------
// Host-side orchestration
// ---------------------------------------------------------------------------
static inline void launch_gemm(bool nk, const __bf16* A, const __bf16* B, const float* bias,
                               float* C, int M, int N, int K, int ldb, hipStream_t s) {
  int tiles = (M / 16) * (N / 64);          // 16x64 panel per wave
  int blocks = (tiles * 32 + 255) / 256;
  if (nk) gemm_bf16_kernel<true ><<<blocks, 256, 0, s>>>(A, B, bias, C, M, N, K, ldb);
  else    gemm_bf16_kernel<false><<<blocks, 256, 0, s>>>(A, B, bias, C, M, N, K, ldb);
}

static inline void launch_conv(const float* src, __bf16* dst, int n, hipStream_t s) {
  f32_to_bf16_kernel<<<(n + 255) / 256, 256, 0, s>>>(src, dst, n);
}

extern "C" void kernel_launch(void* const* d_in, const int* in_sizes, int n_in,
                              void* d_out, int out_size, void* d_ws, size_t ws_size,
                              hipStream_t stream) {
  const float* enc      = (const float*)d_in[0];   // [B,N,E]
  const float* s_h      = (const float*)d_in[1];   // [1,B,S]
  const float* emb      = (const float*)d_in[2];   // [V,EMB]
  const float* W_enc    = (const float*)d_in[3];   // [E,A]
  const float* b_enc    = (const float*)d_in[4];
  const float* W_struct = (const float*)d_in[5];   // [S,A]
  const float* b_struct = (const float*)d_in[6];
  const float* W_cell   = (const float*)d_in[7];   // [H,A]
  const float* b_cell   = (const float*)d_in[8];
  const float* v_att    = (const float*)d_in[9];
  const float* b_att    = (const float*)d_in[10];
  const float* W_ih     = (const float*)d_in[11];  // [3H,IN]
  const float* W_hh     = (const float*)d_in[12];  // [3H,H]
  const float* b_ih     = (const float*)d_in[13];
  const float* b_hh     = (const float*)d_in[14];
  const float* fc_W     = (const float*)d_in[15];  // [V,H]
  const float* fc_b     = (const float*)d_in[16];
  const int*   target   = (const int*)d_in[17];    // [B,T]
  float* out = (float*)d_out;

  // ---- workspace layout (256B aligned) ----
  char* ws = (char*)d_ws;
  size_t off = 0;
  auto alloc = [&](size_t bytes) -> char* {
    char* p = ws + off;
    off += (bytes + 255) & ~(size_t)255;
    return p;
  };
  __bf16* enc_bf     = (__bf16*)alloc((size_t)Bdim * Ndim * Edim * 2);
  __bf16* s_bf       = (__bf16*)alloc((size_t)Bdim * Sdim * 2);
  __bf16* Wenc_bf    = (__bf16*)alloc((size_t)Edim * Adim * 2);
  __bf16* Wstruct_bf = (__bf16*)alloc((size_t)Sdim * Adim * 2);
  __bf16* Wcell_bf   = (__bf16*)alloc((size_t)Hdim * Adim * 2);
  __bf16* Wih_bf     = (__bf16*)alloc((size_t)3 * Hdim * INdim * 2);
  __bf16* Whh_bf     = (__bf16*)alloc((size_t)3 * Hdim * Hdim * 2);
  __bf16* fcW_bf     = (__bf16*)alloc((size_t)Vdim * Hdim * 2);
  float*  encp       = (float*) alloc((size_t)Bdim * Ndim * Adim * 4);
  float*  sproj      = (float*) alloc((size_t)Bdim * Adim * 4);
  float*  cproj      = (float*) alloc((size_t)Bdim * Adim * 4);
  float*  h_f32      = (float*) alloc((size_t)Bdim * Hdim * 4);
  __bf16* h_bf       = (__bf16*)alloc((size_t)Bdim * Hdim * 2);
  __bf16* x_bf       = (__bf16*)alloc((size_t)Bdim * INdim * 2);
  float*  gi         = (float*) alloc((size_t)Bdim * 3 * Hdim * 4);
  float*  gh         = (float*) alloc((size_t)Bdim * 3 * Hdim * 4);
  float*  loss_acc   = (float*) alloc(256);
  (void)ws_size; (void)in_sizes; (void)n_in; (void)out_size;

  // ---- one-time conversions to bf16 (weights stay resident in 192MB L2) ----
  launch_conv(enc,      enc_bf,     Bdim * Ndim * Edim, stream);
  launch_conv(s_h,      s_bf,       Bdim * Sdim,        stream);
  launch_conv(W_enc,    Wenc_bf,    Edim * Adim,        stream);
  launch_conv(W_struct, Wstruct_bf, Sdim * Adim,        stream);
  launch_conv(W_cell,   Wcell_bf,   Hdim * Adim,        stream);
  launch_conv(W_ih,     Wih_bf,     3 * Hdim * INdim,   stream);
  launch_conv(W_hh,     Whh_bf,     3 * Hdim * Hdim,    stream);
  launch_conv(fc_W,     fcW_bf,     Vdim * Hdim,        stream);

  // ---- precompute: enc_proj (+b_enc) and s_proj (+b_struct) ----
  launch_gemm(false, enc_bf, Wenc_bf, b_enc, encp,
              Bdim * Ndim, Adim, Edim, Adim, stream);
  launch_gemm(false, s_bf, Wstruct_bf, b_struct, sproj,
              Bdim, Adim, Sdim, Adim, stream);

  // ---- init h = 0, loss = 0 ----
  init_state_kernel<<<(Bdim * Hdim + 255) / 256, 256, 0, stream>>>(h_f32, h_bf, loss_acc);

  // ---- sequential decode loop ----
  for (int t = 0; t < Tdim; ++t) {
    // c_proj = h @ W_cell + b_cell   [64,256]
    launch_gemm(false, h_bf, Wcell_bf, b_cell, cproj, Bdim, Adim, Hdim, Adim, stream);
    // attention scores + softmax + context -> x[:, :512]
    att_softmax_ctx_kernel<<<Bdim, 256, 0, stream>>>(encp, sproj, cproj, v_att, b_att, enc, x_bf);
    // x[:, 512:768] = embedding[token]
    embed_kernel<<<Bdim, 256, 0, stream>>>(emb, target, t, x_bf);
    // gi = x @ W_ih^T + b_ih ; gh = h @ W_hh^T + b_hh   [64,1536]
    launch_gemm(true, x_bf, Wih_bf, b_ih, gi, Bdim, 3 * Hdim, INdim, INdim, stream);
    launch_gemm(true, h_bf, Whh_bf, b_hh, gh, Bdim, 3 * Hdim, Hdim, Hdim, stream);
    // gates -> h_new (in place)
    gru_elem_kernel<<<(Bdim * Hdim + 255) / 256, 256, 0, stream>>>(gi, gh, h_f32, h_bf);
    // logits = h_new @ fc_W^T + fc_b -> predictions[t]   [64,2048]
    float* logits_t = out + (size_t)t * Bdim * Vdim;
    launch_gemm(true, h_bf, fcW_bf, fc_b, logits_t, Bdim, Vdim, Hdim, Hdim, stream);
    // cross-entropy accumulation
    loss_kernel<<<Bdim, 256, 0, stream>>>(logits_t, target, t, loss_acc);
  }

  // ---- final scalar loss ----
  final_loss_kernel<<<1, 1, 0, stream>>>(loss_acc, out + (size_t)Tdim * Bdim * Vdim);
}
